// SwinTransformer3DBlock_76562087018744
// MI455X (gfx1250) — matrix operand order repaired and checked by reading, not compiled, not run on hardware
//
#include <hip/hip_runtime.h>
#include <math.h>

typedef _Float16 h16;
typedef __attribute__((ext_vector_type(16))) _Float16 v16h;
typedef __attribute__((ext_vector_type(8)))  float    v8f;
typedef __attribute__((ext_vector_type(4)))  unsigned uv4;

union AB { uv4 u[2]; v16h h; };

#define TOK 16128      // 7*48*48
#define CCH 96
#define NWIN 36
#define NTOK 448       // 7*8*8
#define NHEAD 8
#define HDPAD 16

__device__ __forceinline__ float wave_sum(float v) {
    #pragma unroll
    for (int m = 1; m < 32; m <<= 1) v += __shfl_xor(v, m, 32);
    return v;
}

// ---------------- fp32 -> fp16 weight conversion ----------------
__global__ void cvt_kernel(const float* __restrict__ src, h16* __restrict__ dst, int n) {
    int i = blockIdx.x * 256 + threadIdx.x;
    if (i < n) dst[i] = (h16)src[i];
}

// ---------------- patch embed (1x1x1 conv, Cin=2) ----------------
__global__ void patch_kernel(const float* __restrict__ x, const float* __restrict__ pw,
                             const float* __restrict__ pb, float* __restrict__ feat) {
    int i = blockIdx.x * 256 + threadIdx.x;
    if (i >= TOK * CCH) return;
    int t = i / CCH, o = i % CCH;
    feat[i] = x[t] * pw[2*o] + x[TOK + t] * pw[2*o + 1] + pb[o];
}

// ---------------- LayerNorm kernels (one wave per token) ----------------
// mode 0: LN1 + (roll) + window-partition -> fp16
// mode 1: LN2 -> fp16 (raw token order)
// mode 2: final LN -> fp32 out in [C,D,H,W]
template<int MODE>
__global__ __launch_bounds__(256)
void ln_kernel(const float* __restrict__ feat, const float* __restrict__ wn,
               const float* __restrict__ bn, h16* __restrict__ out_h,
               float* __restrict__ out_f, int shifted) {
    int t = blockIdx.x * 8 + (threadIdx.x >> 5);
    int lane = threadIdx.x & 31;
    if (t >= TOK) return;
    const float* row = feat + (size_t)t * CCH;
    float x0 = row[lane], x1 = row[lane + 32], x2 = row[lane + 64];
    float mu = wave_sum(x0 + x1 + x2) * (1.0f / 96.0f);
    float v0 = x0 - mu, v1 = x1 - mu, v2 = x2 - mu;
    float var = wave_sum(v0*v0 + v1*v1 + v2*v2) * (1.0f / 96.0f);
    float rstd = rsqrtf(var + 1e-5f);
    float y0 = v0 * rstd * wn[lane]      + bn[lane];
    float y1 = v1 * rstd * wn[lane + 32] + bn[lane + 32];
    float y2 = v2 * rstd * wn[lane + 64] + bn[lane + 64];
    if constexpr (MODE == 0) {
        int d = t / 2304, h = (t / 48) % 48, w = t % 48;
        if (shifted) { d = (d + 4) % 7; h = (h + 44) % 48; w = (w + 44) % 48; }
        int widx = (h >> 3) * 6 + (w >> 3);
        int n = d * 64 + ((h & 7) << 3) + (w & 7);
        h16* dr = out_h + ((size_t)widx * NTOK + n) * CCH;
        dr[lane] = (h16)y0; dr[lane + 32] = (h16)y1; dr[lane + 64] = (h16)y2;
    } else if constexpr (MODE == 1) {
        h16* dr = out_h + (size_t)t * CCH;
        dr[lane] = (h16)y0; dr[lane + 32] = (h16)y1; dr[lane + 64] = (h16)y2;
    } else {
        out_f[(size_t)lane        * TOK + t] = y0;
        out_f[(size_t)(lane + 32) * TOK + t] = y1;
        out_f[(size_t)(lane + 64) * TOK + t] = y2;
    }
}

// ---------------- generic WMMA GEMM: C[M,N] = A[M,K] @ W[N,K]^T + bias ----------------
// Each wave owns one 16-wide N column and 4 consecutive M-tiles (64 rows):
// all K/32 weight B-fragments are preloaded into registers and reused 4x.
enum { EPI_QKV = 0, EPI_GELU = 1, EPI_PROJ = 2, EPI_FC2 = 3 };

template<int K, int EPI>
__global__ __launch_bounds__(128)
void gemm16_kernel(const h16* __restrict__ A, const h16* __restrict__ W,
                   const float* __restrict__ bias, int M, int N,
                   h16* __restrict__ out_h, float* __restrict__ feat,
                   h16* __restrict__ Qs, h16* __restrict__ Ks, h16* __restrict__ Vt,
                   int shifted) {
    constexpr int KT = K / 32;
    const int lane = threadIdx.x & 31, wv = threadIdx.x >> 5;
    const int ntn = N >> 4;
    const int tile = blockIdx.x * 4 + wv;
    if (tile >= (M >> 6) * ntn) return;
    const int mg = tile / ntn, nt = tile % ntn;
    const int n0 = nt << 4;
    const int lrow = lane & 15, half = lane >> 4;

    // preload all B fragments for this 16-wide output column
    AB bf[KT];
    const h16* brow = W + (size_t)(n0 + lrow) * K;
    #pragma unroll
    for (int kt = 0; kt < KT; ++kt) {
        bf[kt].u[0] = *(const uv4*)(brow + kt * 32 + half * 16);
        bf[kt].u[1] = *(const uv4*)(brow + kt * 32 + half * 16 + 8);
    }
    const int col = n0 + lrow;
    const float bv = bias[col];

    #pragma unroll
    for (int mi = 0; mi < 4; ++mi) {
        const int m0 = (mg << 6) + (mi << 4);
        const h16* arow = A + (size_t)(m0 + lrow) * K;
        v8f acc = {0.f,0.f,0.f,0.f,0.f,0.f,0.f,0.f};
        #pragma unroll
        for (int kt = 0; kt < KT; ++kt) {
            AB a;
            a.u[0] = *(const uv4*)(arow + kt * 32 + half * 16);
            a.u[1] = *(const uv4*)(arow + kt * 32 + half * 16 + 8);
            acc = __builtin_amdgcn_wmma_f32_16x16x32_f16(false, a.h, false, bf[kt].h,
                                                         (short)0, acc, false, false);
        }
        #pragma unroll
        for (int r = 0; r < 8; ++r) {
            const int row = m0 + r + 8 * half;
            float v = acc[r] + bv;
            if constexpr (EPI == EPI_QKV) {
                int s = col / 96, rem = col % 96, head = rem / 12, dim = rem % 12;
                int widx = row / NTOK, n = row % NTOK;
                size_t wh = (size_t)(widx * NHEAD + head);
                if (s == 0)      Qs[(wh * NTOK + n) * HDPAD + dim] = (h16)(v * 0.28867513459481287f);
                else if (s == 1) Ks[(wh * NTOK + n) * HDPAD + dim] = (h16)v;
                else             Vt[(wh * HDPAD + dim) * NTOK + n] = (h16)v;
            } else if constexpr (EPI == EPI_GELU) {
                float g = 0.5f * v * (1.0f + erff(v * 0.7071067811865476f));
                out_h[(size_t)row * N + col] = (h16)g;
            } else if constexpr (EPI == EPI_PROJ) {
                int widx = row / NTOK, n = row % NTOK;
                int dp = n >> 6, hp = ((widx / 6) << 3) + ((n >> 3) & 7), wp = ((widx % 6) << 3) + (n & 7);
                if (shifted) { dp += 3; if (dp >= 7) dp -= 7; hp += 4; if (hp >= 48) hp -= 48; wp += 4; if (wp >= 48) wp -= 48; }
                size_t t = (size_t)dp * 2304 + hp * 48 + wp;
                feat[t * CCH + col] += v;
            } else { // EPI_FC2 (+residual, raw token order)
                feat[(size_t)row * CCH + col] += v;
            }
        }
    }
}

// ---------------- window attention: one wave per (window, head, 16-row block) ----------------
__device__ __forceinline__ int mask_group(int d, int h, int w) {
    int rd = (d < 4) ? 1 : 2;                       // D=7, ws=7, ss=3
    int rh = (h < 40) ? 0 : ((h < 44) ? 1 : 2);     // 48, ws=8, ss=4
    int rw = (w < 40) ? 0 : ((w < 44) ? 1 : 2);
    return rd * 9 + rh * 3 + rw;
}

__global__ __launch_bounds__(64)
void attn_kernel(const h16* __restrict__ Qs, const h16* __restrict__ Ks,
                 const h16* __restrict__ Vt, const float* __restrict__ rpb,
                 h16* __restrict__ attn_out, int shifted) {
    __shared__ float S[2][16][456];
    const int lane = threadIdx.x & 31, wv = threadIdx.x >> 5;
    const int widx = blockIdx.x >> 3, head = blockIdx.x & 7;
    const int m0 = (blockIdx.y * 2 + wv) << 4;
    const int lrow = lane & 15, half = lane >> 4;
    const size_t base = (size_t)(widx * NHEAD + head);
    const int woh = (widx / 6) << 3, wow = (widx % 6) << 3;

    // Q fragment (K padded 12->32 with zeros; dims >=16 zero, 12..15 zero in storage)
    AB qa;
    {
        const h16* qrow = Qs + (base * NTOK + m0 + lrow) * HDPAD;
        qa.u[0] = *(const uv4*)(qrow + half * 8);
        uv4 z = {0u, 0u, 0u, 0u};
        qa.u[1] = z;
    }

    // ---- S = Q K^T + bias (+mask), tile over 28 key blocks ----
    for (int nb = 0; nb < 28; ++nb) {
        AB kb;
        {
            const h16* krow = Ks + (base * NTOK + nb * 16 + lrow) * HDPAD;
            uv4 k0 = *(const uv4*)(krow + half * 8);
            uv4 k1 = *(const uv4*)(krow + 8);
            unsigned mm = 0u;
            k1 &= mm;
            kb.u[0] = k0; kb.u[1] = k1;
        }
        int tn = nb * 16 + lrow;
        int dn = tn >> 6, hn = (tn >> 3) & 7, wn = tn & 7;
        int gk = shifted ? mask_group(dn, woh + hn, wow + wn) : 0;
        v8f c;
        #pragma unroll
        for (int r = 0; r < 8; ++r) {
            int tm = m0 + r + 8 * half;
            int dm = tm >> 6, hm = (tm >> 3) & 7, wm = tm & 7;
            int idx = (dm - dn + 6) * 225 + (hm - hn + 7) * 15 + (wm - wn + 7);
            float b = rpb[idx * NHEAD + head];
            if (shifted && mask_group(dm, woh + hm, wow + wm) != gk) b -= 100.0f;
            c[r] = b;
        }
        c = __builtin_amdgcn_wmma_f32_16x16x32_f16(false, qa.h, false, kb.h,
                                                   (short)0, c, false, false);
        #pragma unroll
        for (int r = 0; r < 8; ++r) S[wv][r + 8 * half][nb * 16 + lrow] = c[r];
    }
    __syncthreads();

    // ---- row stats: lane owns row lrow, cols [half*224, half*224+224) ----
    // keep mxl = rowmax + log(rowsum); softmax fused into the P fragment build.
    float mxl;
    {
        const float* prow = &S[wv][lrow][0];
        int c0 = half * 224;
        float mx = -3.4e38f;
        for (int cc = 0; cc < 224; ++cc) mx = fmaxf(mx, prow[c0 + cc]);
        mx = fmaxf(mx, __shfl_xor(mx, 16, 32));
        float sum = 0.0f;
        for (int cc = 0; cc < 224; ++cc) sum += __expf(prow[c0 + cc] - mx);
        sum += __shfl_xor(sum, 16, 32);
        mxl = mx + __logf(sum);
    }

    // ---- O = P @ V (14 K-chunks of 32 tokens); P = exp(S - mxl) on the fly ----
    v8f o = {0.f,0.f,0.f,0.f,0.f,0.f,0.f,0.f};
    const float* prow = &S[wv][lrow][0];
    const h16* vrow = Vt + (base * HDPAD + lrow) * NTOK;
    for (int kb = 0; kb < 14; ++kb) {
        AB pa, vb;
        int b0 = kb * 32 + half * 8;
        int b1 = kb * 32 + 16 + half * 8;
        #pragma unroll
        for (int e = 0; e < 8; ++e) {
            pa.h[e]     = (h16)__expf(prow[b0 + e] - mxl);
            pa.h[8 + e] = (h16)__expf(prow[b1 + e] - mxl);
        }
        vb.u[0] = *(const uv4*)(vrow + kb * 32 + half * 8);
        vb.u[1] = *(const uv4*)(vrow + kb * 32 + 16 + half * 8);
        o = __builtin_amdgcn_wmma_f32_16x16x32_f16(false, pa.h, false, vb.h,
                                                   (short)0, o, false, false);
    }
    if (lrow < 12) {
        #pragma unroll
        for (int r = 0; r < 8; ++r)
            attn_out[(size_t)(widx * NTOK + m0 + r + 8 * half) * CCH + head * 12 + lrow] = (h16)o[r];
    }
}

// ======================================================================
extern "C" void kernel_launch(void* const* d_in, const int* in_sizes, int n_in,
                              void* d_out, int out_size, void* d_ws, size_t ws_size,
                              hipStream_t stream) {
    const float* x       = (const float*)d_in[0];
    const float* patch_w = (const float*)d_in[1];
    const float* patch_b = (const float*)d_in[2];
    const float* norm1_w = (const float*)d_in[3];
    const float* norm1_b = (const float*)d_in[4];
    const float* qkv_w   = (const float*)d_in[5];
    const float* qkv_b   = (const float*)d_in[6];
    const float* rpb     = (const float*)d_in[7];
    const float* proj_w  = (const float*)d_in[8];
    const float* proj_b  = (const float*)d_in[9];
    const float* norm2_w = (const float*)d_in[10];
    const float* norm2_b = (const float*)d_in[11];
    const float* fc1_w   = (const float*)d_in[12];
    const float* fc1_b   = (const float*)d_in[13];
    const float* fc2_w   = (const float*)d_in[14];
    const float* fc2_b   = (const float*)d_in[15];
    const float* normf_w = (const float*)d_in[16];
    const float* normf_b = (const float*)d_in[17];

    char* ws = (char*)d_ws;
    float* feat  = (float*)(ws + 0);          // 16128*96*4   = 6,193,152
    h16* qkv16   = (h16*)(ws + 6193152);      // 6*288*96*2   =   331,776
    h16* proj16  = (h16*)(ws + 6524928);      // 6*96*96*2    =   110,592
    h16* fc116   = (h16*)(ws + 6635520);      // 6*384*96*2   =   442,368
    h16* fc216   = (h16*)(ws + 7077888);      // 6*96*384*2   =   442,368
    h16* ywin    = (h16*)(ws + 7520256);      // 16128*96*2   = 3,096,576
    h16* aout    = (h16*)(ws + 10616832);     // 16128*96*2   = 3,096,576
    h16* hbuf    = (h16*)(ws + 13713408);     // 16128*384*2  = 12,386,304
    h16* Qs      = (h16*)(ws + 26099712);     // 36*8*448*16*2 = 4,128,768
    h16* Ks      = (h16*)(ws + 30228480);
    h16* Vt      = (h16*)(ws + 34357248);     // end 38,486,016

    // zero Q/K/V once per call so head-dim pad lanes (12..15) stay zero
    hipMemsetAsync(Qs, 0, (size_t)3 * 4128768, stream);

    cvt_kernel<<<(165888 + 255) / 256, 256, 0, stream>>>(qkv_w, qkv16, 165888);
    cvt_kernel<<<(55296  + 255) / 256, 256, 0, stream>>>(proj_w, proj16, 55296);
    cvt_kernel<<<(221184 + 255) / 256, 256, 0, stream>>>(fc1_w, fc116, 221184);
    cvt_kernel<<<(221184 + 255) / 256, 256, 0, stream>>>(fc2_w, fc216, 221184);

    patch_kernel<<<(TOK * CCH + 255) / 256, 256, 0, stream>>>(x, patch_w, patch_b, feat);

    for (int i = 0; i < 6; ++i) {
        const int sh = i & 1;
        ln_kernel<0><<<TOK / 8, 256, 0, stream>>>(feat, norm1_w + i * 96, norm1_b + i * 96,
                                                  ywin, nullptr, sh);
        { int tiles = (TOK / 64) * (288 / 16);
          gemm16_kernel<96, EPI_QKV><<<tiles / 4, 128, 0, stream>>>(
              ywin, qkv16 + (size_t)i * 27648, qkv_b + i * 288, TOK, 288,
              nullptr, nullptr, Qs, Ks, Vt, 0); }
        attn_kernel<<<dim3(NWIN * NHEAD, 14), 64, 0, stream>>>(
            Qs, Ks, Vt, rpb + (size_t)i * 23400, aout, sh);
        { int tiles = (TOK / 64) * (96 / 16);
          gemm16_kernel<96, EPI_PROJ><<<tiles / 4, 128, 0, stream>>>(
              aout, proj16 + (size_t)i * 9216, proj_b + i * 96, TOK, 96,
              nullptr, feat, nullptr, nullptr, nullptr, sh); }
        ln_kernel<1><<<TOK / 8, 256, 0, stream>>>(feat, norm2_w + i * 96, norm2_b + i * 96,
                                                  ywin, nullptr, 0);
        { int tiles = (TOK / 64) * (384 / 16);
          gemm16_kernel<96, EPI_GELU><<<tiles / 4, 128, 0, stream>>>(
              ywin, fc116 + (size_t)i * 36864, fc1_b + i * 384, TOK, 384,
              hbuf, nullptr, nullptr, nullptr, nullptr, 0); }
        { int tiles = (TOK / 64) * (96 / 16);
          gemm16_kernel<384, EPI_FC2><<<tiles / 4, 128, 0, stream>>>(
              hbuf, fc216 + (size_t)i * 36864, fc2_b + i * 96, TOK, 96,
              nullptr, feat, nullptr, nullptr, nullptr, 0); }
    }
    ln_kernel<2><<<TOK / 8, 256, 0, stream>>>(feat, normf_w, normf_b,
                                              nullptr, (float*)d_out, 0);
}
